// Transformer_90855738179903
// MI455X (gfx1250) — matrix operand order repaired and checked by reading, not compiled, not run on hardware
//
#include <hip/hip_runtime.h>

typedef __attribute__((ext_vector_type(16))) __bf16       v16bf;
typedef __attribute__((ext_vector_type(8)))  float        v8f;
typedef __attribute__((ext_vector_type(4)))  unsigned int uvec4;

union FragBF { uvec4 q[2]; v16bf v; };

#define WMMA_BF16(A,B,C) \
  __builtin_amdgcn_wmma_f32_16x16x32_bf16(false,(A),false,(B),(short)0,(C),false,false)

// Issue two async global->LDS b128 copies (32 contiguous bytes/lane slot), ASYNCcnt += 2.
// IOFFSET is added to BOTH the global and LDS addresses (ISA 08_async_tensor §4.4).
#define ASYNC_LD32(ldsOff, gptr)                                            \
  asm volatile("global_load_async_to_lds_b128 %0, %1, off\n\t"              \
               "global_load_async_to_lds_b128 %0, %1, off offset:16"        \
               :: "v"(ldsOff), "v"(gptr) : "memory")

#if defined(__has_builtin)
#if __has_builtin(__builtin_amdgcn_tensor_load_to_lds)
#define HAVE_TDM 1
#endif
#endif
#ifndef HAVE_TDM
#define HAVE_TDM 0
#endif

#if HAVE_TDM
typedef __attribute__((ext_vector_type(4))) unsigned int tdmv4u;
typedef __attribute__((ext_vector_type(8))) int          tdmv8i;
typedef __attribute__((ext_vector_type(4))) int          tdmv4i;

// TDM: load a 128-row x 32-col bf16 tile (row stride = strideElems) into LDS,
// writing rows packed with 16B padding after each 64B row (matches lA/lB layout:
// 40-element row stride). D# built per ISA 08_async_tensor §8.
// amdgpu-toolchain (clang-23) builtin arity: (v4u g0, v8i g1, v4i g2, v4i g3, v8i, i32 cpol).
__device__ __forceinline__ void tdm_load_tile(unsigned lds_addr, const void* gaddr,
                                              unsigned strideElems) {
  unsigned long long ga = (unsigned long long)(size_t)gaddr;
  tdmv4u g0;
  g0[0] = 1u;                                                  // count=1 (valid), no gather
  g0[1] = lds_addr;                                            // LDS byte address
  g0[2] = (unsigned)ga;                                        // global addr [31:0]
  g0[3] = (unsigned)((ga >> 32) & 0x01ffffffu) | (2u << 30);   // addr[56:32] | type=2
  tdmv8i g1;
  g1[0] = (int)((1u << 16)    // data_size = 2 bytes
              | (1u << 20)    // pad_enable
              | (3u << 22)    // pad_interval: 16 DWORDs (64B row)
              | (3u << 25));  // pad_amount:   4 DWORDs (16B pad)
  g1[1] = (int)((strideElems & 0xffffu) << 16);                  // tensor_dim0[15:0]
  g1[2] = (int)(((strideElems >> 16) & 0xffffu) | (128u << 16)); // dim0[31:16] | dim1[15:0]
  g1[3] = (int)(32u << 16);                                      // dim1[31:16]=0 | tile_dim0=32
  g1[4] = (int)128u;                                             // tile_dim1=128, tile_dim2=0
  g1[5] = (int)strideElems;                                      // dim0_stride[31:0]
  g1[6] = 0;                                                     // stride hi / dim1_stride lo
  g1[7] = 0;
  tdmv4i gz  = (tdmv4i)0;
  tdmv8i gz8 = (tdmv8i)0;
  __builtin_amdgcn_tensor_load_to_lds(g0, g1, gz, gz, gz8, 0);
}
#endif

// ---------------- elementwise f32 -> bf16 ----------------
__global__ void k_cvt_bf16(const float* __restrict__ in, __bf16* __restrict__ out, int n) {
  int i = blockIdx.x * 256 + threadIdx.x;
  if (i < n) out[i] = (__bf16)in[i];
}

// ---------------- transpose+convert f32 [R,C] -> bf16 [C,R] ----------------
// grid(C/32, R/32), 256 threads
__global__ void k_transpose_cvt(const float* __restrict__ in, __bf16* __restrict__ out,
                                int R, int C) {
  __shared__ float tile[32][33];
  int tx = threadIdx.x & 31, ty = threadIdx.x >> 5;
  int c0 = blockIdx.x * 32, r0 = blockIdx.y * 32;
  for (int i = 0; i < 4; i++) {
    int r = ty * 4 + i;
    tile[r][tx] = in[(size_t)(r0 + r) * C + c0 + tx];
  }
  __syncthreads();
  for (int i = 0; i < 4; i++) {
    int rr = ty * 4 + i;  // source column index within tile
    out[(size_t)(c0 + rr) * R + r0 + tx] = (__bf16)tile[tx][rr];
  }
}

// ---------------- V [B,T,H*D] bf16 -> V^T [B*H, D, T] bf16 ----------------
// grid.x = (T/32)*(D/32)=256, grid.y = B*H=16, 256 threads
__global__ void k_transpose_v(const __bf16* __restrict__ vb, __bf16* __restrict__ vtb) {
  __shared__ __bf16 tile[32][33];
  int tx = threadIdx.x & 31, ty = threadIdx.x >> 5;
  int bh = blockIdx.y, b = bh >> 2, h = bh & 3;
  int tt = (blockIdx.x & 31) * 32;  // T tile base
  int td = (blockIdx.x >> 5) * 32;  // D tile base
  for (int i = 0; i < 4; i++) {
    int t = tt + ty * 4 + i;
    tile[ty * 4 + i][tx] = vb[((size_t)(b * 1024 + t)) * 1024 + h * 256 + td + tx];
  }
  __syncthreads();
  for (int i = 0; i < 4; i++) {
    int rr = ty * 4 + i;  // d offset within tile
    vtb[((size_t)bh * 256 + td + rr) * 1024 + tt + tx] = tile[tx][rr];
  }
}

// ---------------- tiled bf16 WMMA GEMM: C = (A @ Bt^T + bias) * scale ----------------
// A: [M,K] bf16 row-major, Bt: [N,K] bf16 row-major (B pre-transposed).
// 256 threads (8 waves), 128x128 tile per block, each wave owns 32x64.
// Global->LDS staging: TDM tensor_load_to_lds (TENSORcnt) when available,
// else double-buffered GLOBAL_LOAD_ASYNC_TO_LDS_B128 (ASYNCcnt). grid(N/128, M/128)
__global__ void k_gemm(const __bf16* __restrict__ A, const __bf16* __restrict__ Bt,
                       const float* __restrict__ bias,
                       float* __restrict__ Cf, __bf16* __restrict__ Cb,
                       int M, int N, int K, float scale, int relu) {
  // padded rows: 32 bf16 data + 8 pad -> 80B stride (16B aligned, bank-spread)
  __shared__ __align__(16) __bf16 lA[2][128 * 40];
  __shared__ __align__(16) __bf16 lB[2][128 * 40];
  const unsigned BUFB = 128 * 40 * 2;  // bytes per buffer
  int lane = threadIdx.x & 31, wv = threadIdx.x >> 5;
  int hi = lane >> 4, ln = lane & 15;
  int m0 = blockIdx.y * 128, n0 = blockIdx.x * 128;
  int wr = (wv >> 1) * 32, wc = (wv & 1) * 64;
  v8f acc[2][4];
  for (int a = 0; a < 2; a++)
    for (int c = 0; c < 4; c++)
      for (int r = 0; r < 8; r++) acc[a][c][r] = 0.f;

  int nsteps = K >> 5;
  // low 32 bits of a generic shared pointer == LDS byte offset (aperture mapping)
  unsigned ldsA0 = (unsigned)(size_t)&lA[0][0];
  unsigned ldsB0 = (unsigned)(size_t)&lB[0][0];

#if HAVE_TDM
  bool issuer = (wv == 0);
  const __bf16* gA = A  + (size_t)m0 * K;
  const __bf16* gB = Bt + (size_t)n0 * K;
  if (issuer) {
    tdm_load_tile(ldsA0, gA, (unsigned)K);
    tdm_load_tile(ldsB0, gB, (unsigned)K);
  }
#else
  // each thread stages one 32B slot of A and one of B per K-step
  int lrow = threadIdx.x >> 1, lhalf = threadIdx.x & 1;
  unsigned ldsA = ldsA0 + (unsigned)((lrow * 40 + lhalf * 16) * 2);
  unsigned ldsB = ldsB0 + (unsigned)((lrow * 40 + lhalf * 16) * 2);
  const __bf16* ga = A  + (size_t)(m0 + lrow) * K + lhalf * 16;
  const __bf16* gb = Bt + (size_t)(n0 + lrow) * K + lhalf * 16;
  ASYNC_LD32(ldsA, ga);
  ASYNC_LD32(ldsB, gb);
#endif

  for (int s = 0; s < nsteps; s++) {
    int cur = s & 1;
#if HAVE_TDM
    if (issuer) {
      if (s + 1 < nsteps) {
        unsigned boff = (unsigned)(cur ^ 1) * BUFB;
        tdm_load_tile(ldsA0 + boff, gA + (s + 1) * 32, (unsigned)K);
        tdm_load_tile(ldsB0 + boff, gB + (s + 1) * 32, (unsigned)K);
        // 4 in flight; wait until only the 2 newest remain -> current buffer done
        __builtin_amdgcn_s_wait_tensorcnt(2);
      } else {
        __builtin_amdgcn_s_wait_tensorcnt(0);
      }
    }
#else
    if (s + 1 < nsteps) {
      unsigned boff = (unsigned)(cur ^ 1) * BUFB;
      ASYNC_LD32(ldsA + boff, ga + (s + 1) * 32);
      ASYNC_LD32(ldsB + boff, gb + (s + 1) * 32);
      asm volatile("s_wait_asynccnt 0x4" ::: "memory");
    } else {
      asm volatile("s_wait_asynccnt 0x0" ::: "memory");
    }
#endif
    __syncthreads();  // staged tile visible to all waves
    FragBF fa[2], fb[4];
    for (int f = 0; f < 2; f++) {
      const __bf16* pa = &lA[cur][(wr + f * 16 + ln) * 40];
      fa[f].q[0] = *(const uvec4*)(pa + hi * 8);
      fa[f].q[1] = *(const uvec4*)(pa + hi * 8 + 16);
    }
    for (int f = 0; f < 4; f++) {
      const __bf16* pb = &lB[cur][(wc + f * 16 + ln) * 40 + hi * 8];
      fb[f].q[0] = *(const uvec4*)(pb);
      fb[f].q[1] = *(const uvec4*)(pb + 16);
    }
    for (int a = 0; a < 2; a++)
      for (int c = 0; c < 4; c++)
        acc[a][c] = WMMA_BF16(fa[a].v, fb[c].v, acc[a][c]);
    __syncthreads();  // reads done before this buffer is overwritten next step
  }

  for (int a = 0; a < 2; a++)
    for (int c = 0; c < 4; c++) {
      int n = n0 + wc + c * 16 + ln;
      float bv = bias ? bias[n] : 0.f;
      for (int r = 0; r < 8; r++) {
        int m = m0 + wr + a * 16 + r + 8 * hi;
        float o = (acc[a][c][r] + bv) * scale;
        if (relu) o = fmaxf(o, 0.f);
        size_t idx = (size_t)m * N + n;
        if (Cf) Cf[idx] = o;
        if (Cb) Cb[idx] = (__bf16)o;
      }
    }
}

// ---------------- flash attention (one (b,h), 64 query rows per 128-thread block) ----
// qb/kb: [B,T,H*D] bf16 (q pre-scaled by 1/sqrt(D)); vtb: [B*H, D, T] bf16
// ob: [B,T,H*D] bf16. grid(T/64, B*H).
// K/V tiles staged cooperatively into LDS via async global->LDS (shared by 4 waves).
__global__ void k_attn(const __bf16* __restrict__ qb, const __bf16* __restrict__ kb,
                       const __bf16* __restrict__ vtb, __bf16* __restrict__ ob) {
  __shared__ __align__(16) __bf16 lK[64 * 264];   // 64 keys x 256 d (+8 pad)
  __shared__ __align__(16) __bf16 lV[256 * 72];   // 256 d x 64 t (+8 pad)
  __shared__ __align__(16) __bf16 lp[4][16][72];
  const int HD = 1024, T = 1024, Dh = 256;
  int tid = threadIdx.x;
  int lane = tid & 31, wv = tid >> 5;
  int hi = lane >> 4, ln = lane & 15;
  int bh = blockIdx.y, b = bh >> 2, h = bh & 3;
  int q0 = blockIdx.x * 64 + wv * 16;

  // Cache this wave's Q fragments (16 rows x 256 d) in registers
  const __bf16* qp = qb + ((size_t)(b * T + q0 + ln)) * HD + h * Dh;
  FragBF qf[8];
  for (int s = 0; s < 8; s++) {
    qf[s].q[0] = *(const uvec4*)(qp + s * 32 + hi * 8);
    qf[s].q[1] = *(const uvec4*)(qp + s * 32 + hi * 8 + 16);
  }
  v8f oacc[16];
  for (int f = 0; f < 16; f++)
    for (int r = 0; r < 8; r++) oacc[f][r] = 0.f;
  float rm[8], rs[8];
  for (int r = 0; r < 8; r++) { rm[r] = -3.0e38f; rs[r] = 0.f; }

  const __bf16* kbase = kb + ((size_t)(b * T)) * HD + h * Dh;
  const __bf16* vbase = vtb + (size_t)bh * Dh * T;

  // staging assignments: K: thread -> (row = tid/2, 128-elem half); V: rows 2*tid, 2*tid+1
  int krow = tid >> 1, khalf = tid & 1;
  unsigned ldsK = (unsigned)(size_t)&lK[0] + (unsigned)((krow * 264 + khalf * 128) * 2);
  unsigned ldsV = (unsigned)(size_t)&lV[0] + (unsigned)((2 * tid * 72) * 2);
  const __bf16* gK0 = kbase + (size_t)krow * HD + khalf * 128;
  const __bf16* gV0 = vbase + (size_t)(2 * tid) * T;

  for (int j = 0; j < 16; j++) {  // 64-key blocks
    // cooperative async stage of K (64x256) and V^T (256x64) tiles
    const __bf16* gK = gK0 + (size_t)(j * 64) * HD;
    const __bf16* gV = gV0 + j * 64;
    for (int c = 0; c < 8; c++) ASYNC_LD32(ldsK + c * 32, gK + c * 16);
    for (int c = 0; c < 4; c++) {
      ASYNC_LD32(ldsV + c * 32,       gV + c * 16);
      ASYNC_LD32(ldsV + 144 + c * 32, gV + T + c * 16);
    }
    asm volatile("s_wait_asynccnt 0x0" ::: "memory");
    __syncthreads();  // tiles visible to all waves

    v8f sacc[4];
    for (int f = 0; f < 4; f++)
      for (int r = 0; r < 8; r++) sacc[f][r] = 0.f;
    // scores: S[16,64] += Q[16,32] x K^T[32,16] over 8 K-steps (K frags from LDS)
    for (int s = 0; s < 8; s++)
      for (int f = 0; f < 4; f++) {
        FragBF kf;
        const __bf16* kp = &lK[(f * 16 + ln) * 264 + s * 32 + hi * 8];
        kf.q[0] = *(const uvec4*)kp;
        kf.q[1] = *(const uvec4*)(kp + 16);
        sacc[f] = WMMA_BF16(qf[s].v, kf.v, sacc[f]);
      }
    // online softmax: row max across n (16 lanes of a half-wave)
    float nm[8], al[8];
    for (int r = 0; r < 8; r++) {
      float t = fmaxf(fmaxf(sacc[0][r], sacc[1][r]), fmaxf(sacc[2][r], sacc[3][r]));
      t = fmaxf(t, __shfl_xor(t, 1));
      t = fmaxf(t, __shfl_xor(t, 2));
      t = fmaxf(t, __shfl_xor(t, 4));
      t = fmaxf(t, __shfl_xor(t, 8));
      nm[r] = fmaxf(rm[r], t);
      al[r] = __expf(rm[r] - nm[r]);
      rm[r] = nm[r];
    }
    // P = exp(S - max); spill to LDS (C-layout -> A-layout conversion)
    for (int f = 0; f < 4; f++)
      for (int r = 0; r < 8; r++) {
        float pv = __expf(sacc[f][r] - nm[r]);
        sacc[f][r] = pv;
        lp[wv][r + 8 * hi][f * 16 + ln] = (__bf16)pv;
      }
    for (int r = 0; r < 8; r++) {
      float s2 = sacc[0][r] + sacc[1][r] + sacc[2][r] + sacc[3][r];
      s2 += __shfl_xor(s2, 1);
      s2 += __shfl_xor(s2, 2);
      s2 += __shfl_xor(s2, 4);
      s2 += __shfl_xor(s2, 8);
      rs[r] = rs[r] * al[r] + s2;
    }
    for (int f = 0; f < 16; f++)
      for (int r = 0; r < 8; r++) oacc[f][r] *= al[r];
    asm volatile("s_wait_dscnt 0x0" ::: "memory");  // wave-local LDS RAW fence
    FragBF pf[2];
    for (int s2 = 0; s2 < 2; s2++) {
      const __bf16* pp = &lp[wv][ln][s2 * 32];
      pf[s2].q[0] = *(const uvec4*)(pp + hi * 8);
      pf[s2].q[1] = *(const uvec4*)(pp + hi * 8 + 16);
    }
    // O[16,256] += P[16,32] x V[32,16] per d-block (V frags from LDS)
    for (int s2 = 0; s2 < 2; s2++)
      for (int f = 0; f < 16; f++) {
        FragBF vf;
        const __bf16* vp = &lV[(f * 16 + ln) * 72 + s2 * 32 + hi * 8];
        vf.q[0] = *(const uvec4*)vp;
        vf.q[1] = *(const uvec4*)(vp + 16);
        oacc[f] = WMMA_BF16(pf[s2].v, vf.v, oacc[f]);
      }
    __syncthreads();  // all reads done before next j overwrites lK/lV
  }
  __bf16* op = ob + ((size_t)(b * T)) * HD + h * Dh;
  for (int f = 0; f < 16; f++)
    for (int r = 0; r < 8; r++)
      op[((size_t)(q0 + r + 8 * hi)) * HD + f * 16 + ln] = (__bf16)(oacc[f][r] / rs[r]);
}

// ---------------- residual + layernorm (one wave per 256-wide row) ----------------
// grid(NT/8), 256 threads
__global__ void k_add_ln(const float* __restrict__ A, const float* __restrict__ R,
                         const float* __restrict__ g, const float* __restrict__ bta,
                         float* __restrict__ outF, __bf16* __restrict__ outB) {
  int lane = threadIdx.x & 31, wv = threadIdx.x >> 5;
  int row = blockIdx.x * 8 + wv;
  size_t base = (size_t)row * 256 + lane * 8;
  float v[8];
  for (int i = 0; i < 8; i++) v[i] = A[base + i] + R[base + i];
  float s = 0.f;
  for (int i = 0; i < 8; i++) s += v[i];
  for (int m = 1; m < 32; m <<= 1) s += __shfl_xor(s, m);
  float mean = s * (1.f / 256.f);
  float vs = 0.f;
  for (int i = 0; i < 8; i++) { float d = v[i] - mean; vs += d * d; }
  for (int m = 1; m < 32; m <<= 1) vs += __shfl_xor(vs, m);
  float inv = rsqrtf(vs * (1.f / 256.f) + 1e-5f);
  for (int i = 0; i < 8; i++) {
    int c = lane * 8 + i;
    float o = (v[i] - mean) * inv * g[c] + bta[c];
    outF[base + i] = o;
    if (outB) outB[base + i] = (__bf16)o;
  }
}

extern "C" void kernel_launch(void* const* d_in, const int* in_sizes, int n_in,
                              void* d_out, int out_size, void* d_ws, size_t ws_size,
                              hipStream_t stream) {
  (void)in_sizes; (void)n_in; (void)out_size; (void)ws_size;
  const float* queries = (const float*)d_in[0];
  const float* Wq = (const float*)d_in[1];
  const float* bq = (const float*)d_in[2];
  const float* Wk = (const float*)d_in[3];
  const float* bk = (const float*)d_in[4];
  const float* Wv = (const float*)d_in[5];
  const float* bv = (const float*)d_in[6];
  const float* Wo = (const float*)d_in[7];
  const float* bo = (const float*)d_in[8];
  const float* ln1_s = (const float*)d_in[9];
  const float* ln1_b = (const float*)d_in[10];
  const float* W1 = (const float*)d_in[11];
  const float* b1 = (const float*)d_in[12];
  const float* W2 = (const float*)d_in[13];
  const float* b2 = (const float*)d_in[14];
  const float* ln2_s = (const float*)d_in[15];
  const float* ln2_b = (const float*)d_in[16];

  const int Bb = 4, T = 1024, D = 256, H = 4, M = 1024, HD = 1024;
  const int NT = Bb * T;  // 4096 token rows

  char* p = (char*)d_ws;
  auto alloc = [&](size_t bytes) -> char* {
    char* r = p;
    p += (bytes + 255) & ~(size_t)255;
    return r;
  };
  __bf16* xb  = (__bf16*)alloc((size_t)NT * D * 2);
  __bf16* yb  = (__bf16*)alloc((size_t)NT * D * 2);
  __bf16* qbf = (__bf16*)alloc((size_t)NT * HD * 2);
  __bf16* kbf = (__bf16*)alloc((size_t)NT * HD * 2);
  __bf16* vbf = (__bf16*)alloc((size_t)NT * HD * 2);
  __bf16* vtb = (__bf16*)alloc((size_t)Bb * H * D * T * 2);
  __bf16* obf = (__bf16*)alloc((size_t)NT * HD * 2);
  __bf16* hbf = (__bf16*)alloc((size_t)NT * M * 2);
  float*  t256 = (float*)alloc((size_t)NT * D * 4);
  float*  xws  = (float*)alloc((size_t)NT * D * 4);
  float*  yws  = (float*)alloc((size_t)NT * D * 4);
  __bf16* WqT = (__bf16*)alloc((size_t)D * HD * 2);
  __bf16* WkT = (__bf16*)alloc((size_t)D * HD * 2);
  __bf16* WvT = (__bf16*)alloc((size_t)D * HD * 2);
  __bf16* WoT = (__bf16*)alloc((size_t)HD * D * 2);
  __bf16* W1T = (__bf16*)alloc((size_t)D * M * 2);
  __bf16* W2T = (__bf16*)alloc((size_t)M * D * 2);

  for (int l = 0; l < 2; l++) {
    const float* xin = (l == 0) ? queries : xws;
    // activation -> bf16
    k_cvt_bf16<<<(NT * D + 255) / 256, 256, 0, stream>>>(xin, xb, NT * D);
    // weights -> transposed bf16 [N,K]
    k_transpose_cvt<<<dim3(HD / 32, D / 32), 256, 0, stream>>>(Wq + (size_t)l * D * HD, WqT, D, HD);
    k_transpose_cvt<<<dim3(HD / 32, D / 32), 256, 0, stream>>>(Wk + (size_t)l * D * HD, WkT, D, HD);
    k_transpose_cvt<<<dim3(HD / 32, D / 32), 256, 0, stream>>>(Wv + (size_t)l * D * HD, WvT, D, HD);
    k_transpose_cvt<<<dim3(D / 32, HD / 32), 256, 0, stream>>>(Wo + (size_t)l * HD * D, WoT, HD, D);
    k_transpose_cvt<<<dim3(M / 32, D / 32), 256, 0, stream>>>(W1 + (size_t)l * D * M, W1T, D, M);
    k_transpose_cvt<<<dim3(D / 32, M / 32), 256, 0, stream>>>(W2 + (size_t)l * M * D, W2T, M, D);
    // QKV projections (q pre-scaled by 1/sqrt(D) = 1/16)
    k_gemm<<<dim3(HD / 128, NT / 128), 256, 0, stream>>>(xb, WqT, bq + (size_t)l * HD,
                                                         nullptr, qbf, NT, HD, D, 0.0625f, 0);
    k_gemm<<<dim3(HD / 128, NT / 128), 256, 0, stream>>>(xb, WkT, bk + (size_t)l * HD,
                                                         nullptr, kbf, NT, HD, D, 1.f, 0);
    k_gemm<<<dim3(HD / 128, NT / 128), 256, 0, stream>>>(xb, WvT, bv + (size_t)l * HD,
                                                         nullptr, vbf, NT, HD, D, 1.f, 0);
    k_transpose_v<<<dim3(256, Bb * H), 256, 0, stream>>>(vbf, vtb);
    // attention
    k_attn<<<dim3(T / 64, Bb * H), 128, 0, stream>>>(qbf, kbf, vtb, obf);
    // output projection -> f32
    k_gemm<<<dim3(D / 128, NT / 128), 256, 0, stream>>>(obf, WoT, bo + (size_t)l * D,
                                                        t256, nullptr, NT, D, HD, 1.f, 0);
    // residual + LN1 -> yws (f32) + yb (bf16)
    k_add_ln<<<NT / 8, 256, 0, stream>>>(t256, xin, ln1_s + (size_t)l * D, ln1_b + (size_t)l * D,
                                         yws, yb);
    // MLP
    k_gemm<<<dim3(M / 128, NT / 128), 256, 0, stream>>>(yb, W1T, b1 + (size_t)l * M,
                                                        nullptr, hbf, NT, M, D, 1.f, 1);
    k_gemm<<<dim3(D / 128, NT / 128), 256, 0, stream>>>(hbf, W2T, b2 + (size_t)l * D,
                                                        t256, nullptr, NT, D, M, 1.f, 0);
    // residual + LN2 -> next x (f32); final layer writes d_out
    float* xout = (l == 1) ? (float*)d_out : xws;
    k_add_ln<<<NT / 8, 256, 0, stream>>>(t256, yws, ln2_s + (size_t)l * D, ln2_b + (size_t)l * D,
                                         xout, nullptr);
  }
}
